// BaselineMoE_75110388072960
// MI455X (gfx1250) — matrix operand order repaired and checked by reading, not compiled, not run on hardware
//
#include <hip/hip_runtime.h>
#include <hip/hip_bf16.h>

#define HDIM 2048
#define IDIM 1024
#define NEXP 32
#define NETOT 34
#define NTOK 2048
#define CAP 2048
#define TOPK 6
#define SCALEF 1.0f

#define TM 128
#define TN 64
#define KS 32
#define LDA 48
#define LDB 48

typedef __attribute__((ext_vector_type(16))) __bf16 v16bf;
typedef __attribute__((ext_vector_type(8)))  float  v8f;

#if __has_builtin(__builtin_amdgcn_global_load_async_to_lds_b128) && \
    __has_builtin(__builtin_amdgcn_s_wait_asynccnt)
#define USE_ASYNC 1
#endif

// builtin expects: (int4 AS1*, int4 AS3*, imm offset, imm cpol) with
// GCC-style vector_size(16) int vectors (clang prints AS1 as __device__)
typedef int v4i_b __attribute__((vector_size(16)));
typedef v4i_b __attribute__((address_space(1)))* gas128_t;
typedef v4i_b __attribute__((address_space(3)))* las128_t;

// f32 -> bf16, round-half-up (1 VALU)
__device__ __forceinline__ ushort f2bf(float f) {
  return (ushort)((__float_as_uint(f) + 0x8000u) >> 16);
}

// pack two f32 -> packed bf16 pair (low = a, high = b): 2 adds + 1 v_perm_b32
__device__ __forceinline__ uint pk2bf(float a, float b) {
  uint ua = __float_as_uint(a) + 0x8000u;
  uint ub = __float_as_uint(b) + 0x8000u;
  // sel 0x07060302: d0=a.b2, d1=a.b3 (low dword = 2nd arg), d2=b.b2, d3=b.b3
  return __builtin_amdgcn_perm(ub, ua, 0x07060302u);
}

__device__ __forceinline__ float fastrcp(float x) {
#if __has_builtin(__builtin_amdgcn_rcpf)
  return __builtin_amdgcn_rcpf(x);
#else
  return 1.f / x;
#endif
}

// copy 32 bytes global(bf16) -> LDS, async on CDNA5
__device__ __forceinline__ void cpA32(const ushort* g, ushort* l) {
#ifdef USE_ASYNC
  __builtin_amdgcn_global_load_async_to_lds_b128((gas128_t)g, (las128_t)l, 0, 0);
  __builtin_amdgcn_global_load_async_to_lds_b128((gas128_t)g, (las128_t)l, 16, 0);
#else
  const uint4* s = (const uint4*)g;
  uint4* d = (uint4*)l;
  d[0] = s[0]; d[1] = s[1];
#endif
}
__device__ __forceinline__ void waitAsync() {
#ifdef USE_ASYNC
  __builtin_amdgcn_s_wait_asynccnt(0);
#endif
}

// ---------------- prep: residual init + x -> bf16 ----------------
__global__ void k_prep(const float* __restrict__ x, float* __restrict__ out,
                       ushort* __restrict__ xbf) {
  size_t i = ((size_t)blockIdx.x * blockDim.x + threadIdx.x) * 4;
  float4 v = *(const float4*)(x + i);
  *(float4*)(out + i) = v;
  *(uint2*)(xbf + i) = make_uint2(pk2bf(v.x, v.y), pk2bf(v.z, v.w));
}

// ---------------- router ----------------
__global__ void k_router(const float* __restrict__ x, const float* __restrict__ Wr,
                         int* __restrict__ counts, int* __restrict__ idxl,
                         float* __restrict__ gatel) {
  int lane = threadIdx.x & 31;
  int t = blockIdx.x * (blockDim.x >> 5) + (threadIdx.x >> 5);
  const float* xr = x + (size_t)t * HDIM;
  float acc = 0.f;
  for (int h = 0; h < HDIM; h += 4) {
    float4 xv = *(const float4*)(xr + h);
    acc = fmaf(xv.x, Wr[(h + 0) * NEXP + lane], acc);
    acc = fmaf(xv.y, Wr[(h + 1) * NEXP + lane], acc);
    acc = fmaf(xv.z, Wr[(h + 2) * NEXP + lane], acc);
    acc = fmaf(xv.w, Wr[(h + 3) * NEXP + lane], acc);
  }
  float m = acc;
  for (int o = 16; o; o >>= 1) m = fmaxf(m, __shfl_xor(m, o));
  float p = __expf(acc - m);
  float s = p;
  for (int o = 16; o; o >>= 1) s += __shfl_xor(s, o);
  float wv = p / s;
  for (int k = 0; k < TOPK; ++k) {
    float bm = wv; int ba = lane;
    for (int o = 16; o; o >>= 1) {
      float om = __shfl_xor(bm, o); int oa = __shfl_xor(ba, o);
      if (om > bm || (om == bm && oa < ba)) { bm = om; ba = oa; }
    }
    if (lane == ba) {
      int slot = atomicAdd(&counts[lane], 1);
      idxl[lane * CAP + slot] = t;
      gatel[lane * CAP + slot] = wv * SCALEF;
      wv = -1e30f;
    }
  }
  if (lane == 0) {
    idxl[32 * CAP + t] = t; gatel[32 * CAP + t] = 1.0f;
    idxl[33 * CAP + t] = t; gatel[33 * CAP + t] = 1.0f;
  }
}

// ---------------- scan ----------------
__global__ void k_scan(int* __restrict__ counts, int* __restrict__ offs) {
  if (threadIdx.x == 0) {
    int off = 0;
    for (int e = 0; e < NEXP; ++e) { offs[e] = off; off += counts[e]; }
    offs[32] = off;
    offs[33] = off + NTOK;
    counts[32] = NTOK; counts[33] = NTOK;
  }
}

// ---------------- up: hidden = sigmoid(x@Wg) * (x@Wu) ----------------
__global__ __launch_bounds__(256) void k_up(
    const ushort* __restrict__ xbf,
    const float* __restrict__ Wg, const float* __restrict__ Wu,
    const float* __restrict__ Wg_s, const float* __restrict__ Wu_s,
    const int* __restrict__ counts, const int* __restrict__ offs,
    const int* __restrict__ idxl, ushort* __restrict__ hidden) {
  int e = blockIdx.z;
  int cnt = counts[e];
  int m0 = blockIdx.y * TM;
  if (m0 >= cnt) return;
  int n0 = blockIdx.x * TN;
  const float* wgp = (e < NEXP) ? (Wg + (size_t)e * HDIM * IDIM)
                                : (Wg_s + (size_t)(e - NEXP) * HDIM * IDIM);
  const float* wup = (e < NEXP) ? (Wu + (size_t)e * HDIM * IDIM)
                                : (Wu_s + (size_t)(e - NEXP) * HDIM * IDIM);
  const int* idx = idxl + e * CAP;
  int base = offs[e];

  __shared__ ushort lA[2][TM * LDA];
  __shared__ ushort lB[2][2 * TN * LDB];

  int tid = threadIdx.x;
  int lane = tid & 31, wid = tid >> 5;

  int arow = tid >> 1;
  int acol = (tid & 1) << 4;
  int atok = idx[min(m0 + arow, cnt - 1)];
  const ushort* aSrc = xbf + (size_t)atok * HDIM + acol;
  ushort* aDst0 = &lA[0][arow * LDA + acol];
  ushort* aDst1 = &lA[1][arow * LDA + acol];

  int mat = tid >> 7;
  int q = tid & 127;
  int kb = (q >> 4) << 2;
  int nb = (q & 15) << 2;
  const float* wsel = mat ? wup : wgp;

  v8f zero = {0.f, 0.f, 0.f, 0.f, 0.f, 0.f, 0.f, 0.f};
  v8f cg[4], cu[4];
#pragma unroll
  for (int i = 0; i < 4; ++i) { cg[i] = zero; cu[i] = zero; }

  int half = lane >> 4;
  int mrow = wid * 16 + (lane & 15);
  int kwin = half << 4;

  // prologue: async A(stage0), register-load B(k=0)
  cpA32(aSrc, aDst0);
  float4 b0, b1, b2, b3;
  {
    const float* wp = wsel + (size_t)kb * IDIM + n0 + nb;
    b0 = *(const float4*)(wp);
    b1 = *(const float4*)(wp + IDIM);
    b2 = *(const float4*)(wp + 2 * IDIM);
    b3 = *(const float4*)(wp + 3 * IDIM);
  }

  for (int k0 = 0; k0 < HDIM; k0 += KS) {
    int p = (k0 >> 5) & 1;
    { // store B tile (cvt f32->bf16, transposed [n][k])
      ushort* bb = &lB[p][mat * TN * LDB + nb * LDB + kb];
      *(uint2*)(bb + 0 * LDB) = make_uint2(pk2bf(b0.x, b1.x), pk2bf(b2.x, b3.x));
      *(uint2*)(bb + 1 * LDB) = make_uint2(pk2bf(b0.y, b1.y), pk2bf(b2.y, b3.y));
      *(uint2*)(bb + 2 * LDB) = make_uint2(pk2bf(b0.z, b1.z), pk2bf(b2.z, b3.z));
      *(uint2*)(bb + 3 * LDB) = make_uint2(pk2bf(b0.w, b1.w), pk2bf(b2.w, b3.w));
    }
    waitAsync();          // my async A(stage p) landed
    __syncthreads();      // everyone's stage-p writes visible; frees stage p^1
    if (k0 + KS < HDIM) { // prefetch stage p^1
      cpA32(aSrc + k0 + KS, p ? aDst0 : aDst1);
      const float* wp = wsel + (size_t)(k0 + KS + kb) * IDIM + n0 + nb;
      b0 = *(const float4*)(wp);
      b1 = *(const float4*)(wp + IDIM);
      b2 = *(const float4*)(wp + 2 * IDIM);
      b3 = *(const float4*)(wp + 3 * IDIM);
    }
    // fragments (stage p)
    union { v16bf v; uint u[8]; } af;
#pragma unroll
    for (int v2 = 0; v2 < 8; ++v2) {
      int kk = (half << 4) + (v2 << 1);
      af.u[v2] = *(const uint*)&lA[p][mrow * LDA + kk];
    }
    union BFu { v16bf v; uint4 q2[2]; };
    BFu bg[4], bu[4];
#pragma unroll
    for (int nf = 0; nf < 4; ++nf) {
      int nn = nf * 16 + (lane & 15);
      const uint4* pg = (const uint4*)&lB[p][0 * TN * LDB + nn * LDB + kwin];
      bg[nf].q2[0] = pg[0]; bg[nf].q2[1] = pg[1];
      const uint4* pu = (const uint4*)&lB[p][1 * TN * LDB + nn * LDB + kwin];
      bu[nf].q2[0] = pu[0]; bu[nf].q2[1] = pu[1];
    }
#pragma unroll
    for (int nf = 0; nf < 4; ++nf) {
      cg[nf] = __builtin_amdgcn_wmma_f32_16x16x32_bf16(false, af.v, false, bg[nf].v,
                                                       (short)0, cg[nf], false, false);
      cu[nf] = __builtin_amdgcn_wmma_f32_16x16x32_bf16(false, af.v, false, bu[nf].v,
                                                       (short)0, cu[nf], false, false);
    }
  }
  // epilogue: sigmoid gate via fast rcp, store bf16 hidden
#pragma unroll
  for (int nf = 0; nf < 4; ++nf) {
    int col = n0 + nf * 16 + (lane & 15);
#pragma unroll
    for (int r = 0; r < 8; ++r) {
      int row = wid * 16 + half * 8 + r;
      if (m0 + row < cnt) {
        float g = cg[nf][r], u = cu[nf][r];
        float hv = u * fastrcp(1.f + __expf(-g));
        hidden[(size_t)(base + m0 + row) * IDIM + col] = f2bf(hv);
      }
    }
  }
}

// ---------------- down: out += gate * (hidden @ Wd) ----------------
__global__ __launch_bounds__(256) void k_down(
    const ushort* __restrict__ hidden,
    const float* __restrict__ Wd, const float* __restrict__ Wd_s,
    const int* __restrict__ counts, const int* __restrict__ offs,
    const int* __restrict__ idxl, const float* __restrict__ gatel,
    float* __restrict__ out) {
  int e = blockIdx.z;
  int cnt = counts[e];
  int m0 = blockIdx.y * TM;
  if (m0 >= cnt) return;
  int n0 = blockIdx.x * TN;
  const float* wdp = (e < NEXP) ? (Wd + (size_t)e * IDIM * HDIM)
                                : (Wd_s + (size_t)(e - NEXP) * IDIM * HDIM);
  const int* idx = idxl + e * CAP;
  const float* gl = gatel + e * CAP;
  int base = offs[e];

  __shared__ ushort lA[2][TM * LDA];
  __shared__ ushort lB[2][TN * LDB];

  int tid = threadIdx.x;
  int lane = tid & 31, wid = tid >> 5;

  int arow = tid >> 1;
  int acol = (tid & 1) << 4;
  int aslot = base + min(m0 + arow, cnt - 1);
  const ushort* aSrc = hidden + (size_t)aslot * IDIM + acol;
  ushort* aDst0 = &lA[0][arow * LDA + acol];
  ushort* aDst1 = &lA[1][arow * LDA + acol];

  int kb = (tid >> 4) << 1;   // 0..30
  int nb = (tid & 15) << 2;   // 0..60

  v8f zero = {0.f, 0.f, 0.f, 0.f, 0.f, 0.f, 0.f, 0.f};
  v8f cc[4];
#pragma unroll
  for (int i = 0; i < 4; ++i) cc[i] = zero;

  int half = lane >> 4;
  int mrow = wid * 16 + (lane & 15);
  int kwin = half << 4;

  cpA32(aSrc, aDst0);
  float4 b0, b1;
  {
    const float* wp = wdp + (size_t)kb * HDIM + n0 + nb;
    b0 = *(const float4*)(wp);
    b1 = *(const float4*)(wp + HDIM);
  }

  for (int k0 = 0; k0 < IDIM; k0 += KS) {
    int p = (k0 >> 5) & 1;
    {
      ushort* bb = &lB[p][nb * LDB + kb];
      *(uint*)(bb + 0 * LDB) = pk2bf(b0.x, b1.x);
      *(uint*)(bb + 1 * LDB) = pk2bf(b0.y, b1.y);
      *(uint*)(bb + 2 * LDB) = pk2bf(b0.z, b1.z);
      *(uint*)(bb + 3 * LDB) = pk2bf(b0.w, b1.w);
    }
    waitAsync();
    __syncthreads();
    if (k0 + KS < IDIM) {
      cpA32(aSrc + k0 + KS, p ? aDst0 : aDst1);
      const float* wp = wdp + (size_t)(k0 + KS + kb) * HDIM + n0 + nb;
      b0 = *(const float4*)(wp);
      b1 = *(const float4*)(wp + HDIM);
    }
    union { v16bf v; uint u[8]; } af;
#pragma unroll
    for (int v2 = 0; v2 < 8; ++v2) {
      int kk = (half << 4) + (v2 << 1);
      af.u[v2] = *(const uint*)&lA[p][mrow * LDA + kk];
    }
    union BFu { v16bf v; uint4 q2[2]; };
    BFu bf[4];
#pragma unroll
    for (int nf = 0; nf < 4; ++nf) {
      int nn = nf * 16 + (lane & 15);
      const uint4* pb = (const uint4*)&lB[p][nn * LDB + kwin];
      bf[nf].q2[0] = pb[0]; bf[nf].q2[1] = pb[1];
    }
#pragma unroll
    for (int nf = 0; nf < 4; ++nf) {
      cc[nf] = __builtin_amdgcn_wmma_f32_16x16x32_bf16(false, af.v, false, bf[nf].v,
                                                       (short)0, cc[nf], false, false);
    }
  }
#pragma unroll
  for (int nf = 0; nf < 4; ++nf) {
    int col = n0 + nf * 16 + (lane & 15);
#pragma unroll
    for (int r = 0; r < 8; ++r) {
      int row = wid * 16 + half * 8 + r;
      int mr = m0 + row;
      if (mr < cnt) {
        float gt = gl[mr];
        int tok = idx[mr];
        atomicAdd(&out[(size_t)tok * HDIM + col], cc[nf][r] * gt);
      }
    }
  }
}

// ---------------- host ----------------
extern "C" void kernel_launch(void* const* d_in, const int* in_sizes, int n_in,
                              void* d_out, int out_size, void* d_ws, size_t ws_size,
                              hipStream_t stream) {
  const float* x    = (const float*)d_in[0];
  const float* Wr   = (const float*)d_in[1];
  const float* Wg_s = (const float*)d_in[2];
  const float* Wu_s = (const float*)d_in[3];
  const float* Wd_s = (const float*)d_in[4];
  const float* Wg   = (const float*)d_in[5];
  const float* Wu   = (const float*)d_in[6];
  const float* Wd   = (const float*)d_in[7];
  float* out = (float*)d_out;

  static const size_t CNT_OFF  = 0;
  static const size_t OFF_OFF  = 256;
  static const size_t IDX_OFF  = 512;
  static const size_t GATE_OFF = IDX_OFF + (size_t)NETOT * CAP * 4;
  static const size_t XBF_OFF  = (GATE_OFF + (size_t)NETOT * CAP * 4 + 4095) & ~(size_t)4095;
  static const size_t HID_OFF  = XBF_OFF + (size_t)NTOK * HDIM * 2;

  char* ws = (char*)d_ws;
  int*    counts = (int*)(ws + CNT_OFF);
  int*    offs   = (int*)(ws + OFF_OFF);
  int*    idxl   = (int*)(ws + IDX_OFF);
  float*  gatel  = (float*)(ws + GATE_OFF);
  ushort* xbf    = (ushort*)(ws + XBF_OFF);
  ushort* hid    = (ushort*)(ws + HID_OFF);

  (void)hipMemsetAsync(counts, 0, NETOT * sizeof(int), stream);
  k_prep  <<<(NTOK * HDIM) / 1024, 256, 0, stream>>>(x, out, xbf);
  k_router<<<NTOK / 4, 128, 0, stream>>>(x, Wr, counts, idxl, gatel);
  k_scan  <<<1, 32, 0, stream>>>(counts, offs);
  k_up    <<<dim3(IDIM / TN, NTOK / TM, NETOT), 256, 0, stream>>>(
      xbf, Wg, Wu, Wg_s, Wu_s, counts, offs, idxl, hid);
  k_down  <<<dim3(HDIM / TN, NTOK / TM, NETOT), 256, 0, stream>>>(
      hid, Wd, Wd_s, counts, offs, idxl, gatel, out);
}
